// PDEModelV3_48292612276772
// MI455X (gfx1250) — compile-verified, matrix operand order checked by reading, not run on hardware
//
#include <hip/hip_runtime.h>
#include <hip/hip_bf16.h>

// ---------------- problem constants (match reference) ----------------
constexpr int Dm      = 384;
constexpr int NHEADS  = 12;
constexpr int HD      = 32;          // head dim
constexpr int FFNDIM  = 4 * Dm;      // 1536
constexpr int NLAYERS = 4;
constexpr int Tt      = 8;
constexpr int NHh     = 32;
constexpr int NWw     = 32;
constexpr int Ltot    = Tt * NHh * NWw;   // 8192 tokens
constexpr int KSZ     = 7;

typedef __attribute__((ext_vector_type(16))) _Float16 v16h;
typedef __attribute__((ext_vector_type(8)))  _Float16 v8h;
typedef __attribute__((ext_vector_type(8)))  float    v8f;

__device__ __forceinline__ unsigned lds_off_of(const void* p)
{
    // generic LDS pointer = {aperture_hi32, lds_offset}; low 32 bits are the
    // LDS byte address used by async-to-LDS VDST.
    return (unsigned)(unsigned long long)p;
}

// ---------------------------------------------------------------------
// WMMA GEMM:  C[M,N] = A[M,K](f16) * Bt[N,K](f16, pre-transposed)
//             (+bias)(+residual)
// Block: 256 threads = 8 waves. Each wave owns a 16x64 C strip
// (4 accumulators); block covers 128 rows x 64 cols.
// B tiles (64 cols x 32 k, 4KB each) staged global->LDS with
// GLOBAL_LOAD_ASYNC_TO_LDS_B128, ping-pong double buffered so the async
// transfer of tile i+1 overlaps the 4 WMMAs of tile i
// (wait ASYNCcnt<=1 instead of 0). Fragments read via ds_load_b128.
// ---------------------------------------------------------------------
__global__ __launch_bounds__(256)
void natt_gemm_wmma(const _Float16* __restrict__ A, const _Float16* __restrict__ Bt,
                    const float* __restrict__ bias, const float* __restrict__ residual,
                    float* __restrict__ C, int M, int N, int K)
{
    __shared__ _Float16 Bs[2][64 * 32];    // ping-pong [col][k] tiles, 8 KB

    const int tid   = threadIdx.x;
    const int lane  = tid & 31;
    const int wave  = tid >> 5;
    const int n0    = blockIdx.x * 64;
    const int mtile = blockIdx.y * 8 + wave;

    const int row = lane & 15;             // M row (A) / N col (B) within tile
    const int kb  = (lane >> 4) << 3;      // 0 or 8 (ISA 16-bit fragment layout)

    // staging assignment: this thread copies 16B of Bt into LDS per tile
    const int snn = tid >> 2;              // 0..63  (column within block strip)
    const int sj  = (tid & 3) * 8;         // 0,8,16,24 (k offset in halves)
    const unsigned sdst0 = lds_off_of(&Bs[0][snn * 32 + sj]);
    const unsigned sdst1 = lds_off_of(&Bs[1][snn * 32 + sj]);
    const _Float16* ssrc = Bt + (size_t)(n0 + snn) * K + sj;

    const _Float16* Arow = A + (size_t)(mtile * 16 + row) * K;

    v8f acc[4] = {};
    const int nk = K >> 5;                 // K/32 steps

    // prologue: stage tile 0 into buffer 0
    asm volatile("global_load_async_to_lds_b128 %0, %1, off"
                 :: "v"(sdst0), "v"((unsigned long long)ssrc)
                 : "memory");

    for (int ki = 0; ki < nk; ++ki) {
        const int k0 = ki << 5;

        if (ki + 1 < nk) {
            // stage tile ki+1 into the other buffer, then require only
            // tile ki to be complete (ASYNCcnt <= 1): overlap DMA w/ math
            const unsigned d = (ki & 1) ? sdst0 : sdst1;
            asm volatile("global_load_async_to_lds_b128 %0, %1, off"
                         :: "v"(d), "v"((unsigned long long)(ssrc + k0 + 32))
                         : "memory");
            asm volatile("s_wait_asynccnt 0x1" ::: "memory");
        } else {
            asm volatile("s_wait_asynccnt 0x0" ::: "memory");
        }
        __syncthreads();                   // tile ki visible to all waves

        // ---- A fragment: two 16B global loads per lane ----
        v8h alo = *(const v8h*)(Arow + k0 + kb);
        v8h ahi = *(const v8h*)(Arow + k0 + 16 + kb);
        if (k0 + 32 < K) __builtin_prefetch(Arow + k0 + 32, 0, 1);  // global_prefetch_b8
        v16h a;
        #pragma unroll
        for (int i = 0; i < 8; ++i) { a[i] = alo[i]; a[i + 8] = ahi[i]; }

        // ---- 4 B fragments from LDS (two ds_load_b128 each) + 4 WMMAs ----
        const _Float16* Bbuf = Bs[ki & 1];
        #pragma unroll
        for (int ns = 0; ns < 4; ++ns) {
            const _Float16* bp = Bbuf + (ns * 16 + row) * 32;
            v8h blo = *(const v8h*)(bp + kb);
            v8h bhi = *(const v8h*)(bp + 16 + kb);
            v16h b;
            #pragma unroll
            for (int i = 0; i < 8; ++i) { b[i] = blo[i]; b[i + 8] = bhi[i]; }
            acc[ns] = __builtin_amdgcn_wmma_f32_16x16x32_f16(
                          false, a, false, b, (short)0, acc[ns], false, false);
        }
        __syncthreads();   // reads of buf[ki&1] done before it is re-staged
    }

    // C/D layout: VGPR j -> row (lane<16 ? j : 8+j), col lane&15
    const int col   = lane & 15;
    const int rbase = (lane >> 4) * 8;
    #pragma unroll
    for (int ns = 0; ns < 4; ++ns) {
        const int n = n0 + ns * 16 + col;
        #pragma unroll
        for (int j = 0; j < 8; ++j) {
            const int m = mtile * 16 + rbase + j;
            float v = acc[ns][j];
            if (bias)     v += bias[n];
            if (residual) v += residual[(size_t)m * N + n];
            C[(size_t)m * N + n] = v;
        }
    }
}

// ---------------------------------------------------------------------
// RMSNorm over D=384; one block (128 threads) per token. Optional f16 and
// f32 outputs (f16 feeds the next GEMM's A operand).
// ---------------------------------------------------------------------
__global__ __launch_bounds__(128)
void natt_rmsnorm(const float* __restrict__ x, const float* __restrict__ w,
                  _Float16* __restrict__ out16, float* __restrict__ out32)
{
    __shared__ float red[4];
    const int token = blockIdx.x;
    const float* xr = x + (size_t)token * Dm;

    float vals[3];
    float ss = 0.f;
    #pragma unroll
    for (int i = 0; i < 3; ++i) {
        float v = xr[threadIdx.x + i * 128];
        vals[i] = v;
        ss += v * v;
    }
    #pragma unroll
    for (int off = 16; off > 0; off >>= 1) ss += __shfl_xor(ss, off, 32);
    const int lane = threadIdx.x & 31, wv = threadIdx.x >> 5;
    if (lane == 0) red[wv] = ss;
    __syncthreads();
    const float inv = rsqrtf((red[0] + red[1] + red[2] + red[3]) / (float)Dm + 1e-6f);

    #pragma unroll
    for (int i = 0; i < 3; ++i) {
        int c = threadIdx.x + i * 128;
        float o = vals[i] * inv * w[c];
        if (out16) out16[(size_t)token * Dm + c] = (_Float16)o;
        if (out32) out32[(size_t)token * Dm + c] = o;
    }
}

// ---------------------------------------------------------------------
// 3D neighborhood attention (7x7x7, causal in T, clamped-centered in H/W).
// One wave32 per (token, head); lane = head-dim channel. Online softmax.
// qkv: [L, 3*D] fp32 ; out: [L, D] f16 (feeds proj GEMM).
// ---------------------------------------------------------------------
__global__ __launch_bounds__(256)
void natt_attention(const float* __restrict__ qkv, _Float16* __restrict__ out)
{
    const int lane = threadIdx.x & 31;
    const int wave = threadIdx.x >> 5;
    const int gid  = blockIdx.x * 8 + wave;          // token*NHEADS + head
    const int token = gid / NHEADS;
    const int head  = gid - token * NHEADS;

    const int t  = token >> 10;          // /(32*32)
    const int y  = (token >> 5) & 31;
    const int xw = token & 31;

    const int hd0 = head * HD + lane;
    const float q = qkv[(size_t)token * (3 * Dm) + hd0] * 0.17677669529663687f; // 32^-0.5

    int t0 = t - (KSZ - 1); if (t0 < 0) t0 = 0;                  // causal T
    int ys = y - 3;  if (ys < 0) ys = 0; if (ys > NHh - KSZ) ys = NHh - KSZ;
    int xs = xw - 3; if (xs < 0) xs = 0; if (xs > NWw - KSZ) xs = NWw - KSZ;

    float mrun = -3.0e38f, lrun = 0.f, acc = 0.f;

    for (int tt = t0; tt <= t; ++tt) {
        for (int yy = 0; yy < KSZ; ++yy) {
            for (int xx = 0; xx < KSZ; ++xx) {
                const int ntk = (tt << 10) + ((ys + yy) << 5) + (xs + xx);
                const size_t nb = (size_t)ntk * (3 * Dm);
                float s = q * qkv[nb + Dm + hd0];
                #pragma unroll
                for (int off = 16; off > 0; off >>= 1) s += __shfl_xor(s, off, 32);
                const float vv = qkv[nb + 2 * Dm + hd0];
                const float mn  = fmaxf(mrun, s);
                const float cor = __expf(mrun - mn);
                const float p   = __expf(s - mn);
                lrun = lrun * cor + p;
                acc  = acc  * cor + p * vv;
                mrun = mn;
            }
        }
    }
    out[(size_t)token * Dm + head * HD + lane] = (_Float16)(acc / lrun);
}

// ---------------------------------------------------------------------
__global__ void natt_swiglu(const float* __restrict__ g, const float* __restrict__ u,
                            _Float16* __restrict__ out, int n)
{
    int i = blockIdx.x * blockDim.x + threadIdx.x;
    if (i < n) {
        float gv = g[i];
        float s  = gv / (1.f + __expf(-gv));    // silu
        out[i] = (_Float16)(s * u[i]);
    }
}

// W[K,N] fp32 -> Wt[N,K] f16 (transpose so B-tile columns are contiguous)
__global__ void natt_w_to_f16T(const float* __restrict__ W, _Float16* __restrict__ Wt,
                               int K, int N)
{
    int i = blockIdx.x * blockDim.x + threadIdx.x;      // over N*K
    if (i < N * K) {
        int n = i / K;
        int k = i - n * K;
        Wt[i] = (_Float16)W[(size_t)k * N + n];
    }
}

// ---------------------------------------------------------------------
extern "C" void kernel_launch(void* const* d_in, const int* in_sizes, int n_in,
                              void* d_out, int out_size, void* d_ws, size_t ws_size,
                              hipStream_t stream)
{
    (void)in_sizes; (void)n_in; (void)out_size; (void)ws_size;

    const float* x       = (const float*)d_in[0];
    const float* norm1_w = (const float*)d_in[1];
    const float* norm2_w = (const float*)d_in[2];
    const float* qkv_w   = (const float*)d_in[3];
    const float* qkv_b   = (const float*)d_in[4];
    const float* proj_w  = (const float*)d_in[5];
    const float* proj_b  = (const float*)d_in[6];
    const float* ffn_g   = (const float*)d_in[7];
    const float* ffn_u   = (const float*)d_in[8];
    const float* ffn_d   = (const float*)d_in[9];
    const float* fnorm_w = (const float*)d_in[10];
    float* out = (float*)d_out;

    // ---- workspace layout ----
    char* ws = (char*)d_ws;
    size_t off = 0;
    auto alloc = [&](size_t bytes) -> void* {
        void* p = ws + off;
        off += (bytes + 255) & ~(size_t)255;
        return p;
    };
    float*    h      = (float*)   alloc((size_t)Ltot * Dm * 4);
    _Float16* xn16   = (_Float16*)alloc((size_t)Ltot * Dm * 2);
    float*    qkvbuf = (float*)   alloc((size_t)Ltot * 3 * Dm * 4);
    _Float16* att16  = (_Float16*)alloc((size_t)Ltot * Dm * 2);
    float*    gbuf   = (float*)   alloc((size_t)Ltot * FFNDIM * 4);
    float*    ubuf   = (float*)   alloc((size_t)Ltot * FFNDIM * 4);
    _Float16* mid16  = (_Float16*)alloc((size_t)Ltot * FFNDIM * 2);
    _Float16* qkvwT  = (_Float16*)alloc((size_t)NLAYERS * Dm * 3 * Dm * 2);
    _Float16* projwT = (_Float16*)alloc((size_t)NLAYERS * Dm * Dm * 2);
    _Float16* gatewT = (_Float16*)alloc((size_t)NLAYERS * Dm * FFNDIM * 2);
    _Float16* upwT   = (_Float16*)alloc((size_t)NLAYERS * Dm * FFNDIM * 2);
    _Float16* downwT = (_Float16*)alloc((size_t)NLAYERS * FFNDIM * Dm * 2);

    // ---- per-call weight conversion: fp32 [K,N] -> f16 [N,K] ----
    auto cvtT = [&](const float* src, _Float16* dst, int K, int N) {
        int n = K * N;
        natt_w_to_f16T<<<(n + 255) / 256, 256, 0, stream>>>(src, dst, K, N);
    };
    for (int l = 0; l < NLAYERS; ++l) {
        cvtT(qkv_w  + (size_t)l * Dm * 3 * Dm, qkvwT  + (size_t)l * Dm * 3 * Dm, Dm, 3 * Dm);
        cvtT(proj_w + (size_t)l * Dm * Dm,     projwT + (size_t)l * Dm * Dm,     Dm, Dm);
        cvtT(ffn_g  + (size_t)l * Dm * FFNDIM, gatewT + (size_t)l * Dm * FFNDIM, Dm, FFNDIM);
        cvtT(ffn_u  + (size_t)l * Dm * FFNDIM, upwT   + (size_t)l * Dm * FFNDIM, Dm, FFNDIM);
        cvtT(ffn_d  + (size_t)l * FFNDIM * Dm, downwT + (size_t)l * FFNDIM * Dm, FFNDIM, Dm);
    }

    // residual stream starts as x
    hipMemcpyAsync(h, x, (size_t)Ltot * Dm * 4, hipMemcpyDeviceToDevice, stream);

    const dim3 blk(256);
    const int  mblocks = Ltot / 128;   // 8 M-tiles (16 rows each) per block

    for (int l = 0; l < NLAYERS; ++l) {
        // x_n = rms(h, norm1) -> f16
        natt_rmsnorm<<<Ltot, 128, 0, stream>>>(h, norm1_w + (size_t)l * Dm, xn16, nullptr);

        // qkv = x_n @ qkv_w + qkv_b    [L, 1152] fp32
        natt_gemm_wmma<<<dim3((3 * Dm) / 64, mblocks), blk, 0, stream>>>(
            xn16, qkvwT + (size_t)l * Dm * 3 * Dm,
            qkv_b + (size_t)l * 3 * Dm, nullptr, qkvbuf, Ltot, 3 * Dm, Dm);

        // neighborhood attention -> f16
        natt_attention<<<(Ltot * NHEADS) / 8, blk, 0, stream>>>(qkvbuf, att16);

        // h = h + attn @ proj_w + proj_b   (residual fused, in-place)
        natt_gemm_wmma<<<dim3(Dm / 64, mblocks), blk, 0, stream>>>(
            att16, projwT + (size_t)l * Dm * Dm,
            proj_b + (size_t)l * Dm, h, h, Ltot, Dm, Dm);

        // x_n = rms(h, norm2) -> f16
        natt_rmsnorm<<<Ltot, 128, 0, stream>>>(h, norm2_w + (size_t)l * Dm, xn16, nullptr);

        // gate / up GEMMs
        natt_gemm_wmma<<<dim3(FFNDIM / 64, mblocks), blk, 0, stream>>>(
            xn16, gatewT + (size_t)l * Dm * FFNDIM, nullptr, nullptr, gbuf, Ltot, FFNDIM, Dm);
        natt_gemm_wmma<<<dim3(FFNDIM / 64, mblocks), blk, 0, stream>>>(
            xn16, upwT + (size_t)l * Dm * FFNDIM, nullptr, nullptr, ubuf, Ltot, FFNDIM, Dm);

        // mid = silu(g) * u -> f16
        natt_swiglu<<<(Ltot * FFNDIM + 255) / 256, 256, 0, stream>>>(
            gbuf, ubuf, mid16, Ltot * FFNDIM);

        // h = h + mid @ down_w   (residual fused, in-place)
        natt_gemm_wmma<<<dim3(Dm / 64, mblocks), blk, 0, stream>>>(
            mid16, downwT + (size_t)l * FFNDIM * Dm, nullptr, h, h, Ltot, Dm, FFNDIM);
    }

    // final rmsnorm -> fp32 output
    natt_rmsnorm<<<Ltot, 128, 0, stream>>>(h, fnorm_w, nullptr, out);
}
